// Encoder_74466142978473
// MI455X (gfx1250) — compile-verified, matrix operand order checked by reading, not run on hardware
//
#include <hip/hip_runtime.h>
#include <hip/hip_bf16.h>

// Problem constants (match reference)
#define Bz   8
#define Sz   512
#define Dz   512
#define Hz   8
#define DKz  64
#define DVz  64
#define DFFz 2048
#define Lz   6

typedef __attribute__((ext_vector_type(16))) __bf16 v16bf;
typedef __attribute__((ext_vector_type(8)))  __bf16 v8bf;
typedef __attribute__((ext_vector_type(8)))  float  v8f;

// 16-bit WMMA A/B fragment layout (ISA 7.12.2): per lane (row = lane&15,
// half = lane>>4) the 16 elements are two contiguous 8-element K-runs:
//   elements 0..7  -> K = half*8      .. half*8+7
//   elements 8..15 -> K = 16+half*8   .. 16+half*8+7
// So a fragment is exactly two 16-byte loads from a row-major K-run.
__device__ __forceinline__ v16bf frag_combine(v8bf lo, v8bf hi) {
  v16bf r;
#pragma unroll
  for (int j = 0; j < 8; ++j) { r[j] = lo[j]; r[8 + j] = hi[j]; }
  return r;
}

// Generic bf16 WMMA GEMM: C[M,N] (f32) = alpha * A[M,K] * B
//   BT=1: B given transposed (N x K row-major)
//   BT=0: B given K x N row-major
// Block: 256 threads = 8 wave32 waves. Block tile 256(M) x 64(N);
// wave tile 32 x 64 (2 A-frags x 4 B-frags -> 8 WMMA per K-step).
// B tile (64n x 32k) staged in LDS once per block per K-step.
// grid = (N/64, M/256, batches); batch z decomposed as (z/innerCount, z%innerCount).
template<int BT>
__global__ __launch_bounds__(256) void gemm_bf16_wmma(
    const __bf16* __restrict__ A, const __bf16* __restrict__ Bm, float* __restrict__ C,
    int K, int lda, int ldb, int ldc, int innerCount,
    long long soA, long long siA, long long soB, long long siB,
    long long soC, long long siC, float alpha)
{
  const int z  = blockIdx.z;
  const int z2 = z / innerCount;
  const int z1 = z - z2 * innerCount;
  A  += z2 * soA + z1 * siA;
  Bm += z2 * soB + z1 * siB;
  C  += z2 * soC + z1 * siC;

  const int wave = threadIdx.x >> 5;
  const int lane = threadIdx.x & 31;
  const int r    = lane & 15;   // row within 16 (M for A, N for B/C)
  const int hf   = lane >> 4;   // lane half selects K sub-run / M offset in C
  const int m0   = blockIdx.y * 256 + wave * 32;
  const int n0   = blockIdx.x * 64;

  // LDS B tile: 64 N-rows x 32 K, padded to 40 elems (80B, 16B-aligned rows)
  __shared__ __bf16 bs[64][40];

  v8f acc[2][4] = {};

  for (int k0 = 0; k0 < K; k0 += 32) {
    __syncthreads();
    // ---- cooperative stage of B tile into LDS as [n][k] ----
    if (BT) {
      // B is N x K row-major: thread -> (n, 8-elem k-chunk), 16B global load
      const int nn = threadIdx.x >> 2;          // 0..63
      const int kk = (threadIdx.x & 3) * 8;     // 0,8,16,24
      v8bf t = *(const v8bf*)(Bm + (size_t)(n0 + nn) * ldb + k0 + kk);
      *(v8bf*)&bs[nn][kk] = t;
    } else {
      // B is K x N row-major: thread -> (k, 8-elem n-chunk), 16B global load,
      // scattered 2B LDS stores (transpose into [n][k])
      const int kk = threadIdx.x >> 3;          // 0..31
      const int nn = (threadIdx.x & 7) * 8;     // 0..56
      v8bf t = *(const v8bf*)(Bm + (size_t)(k0 + kk) * ldb + n0 + nn);
#pragma unroll
      for (int j = 0; j < 8; ++j) bs[nn + j][kk] = t[j];
    }
    __syncthreads();

    // ---- A fragments (two 16-row tiles), direct from global, 16B loads ----
    const __bf16* ap0 = A + (size_t)(m0 + r) * lda + k0;
    const __bf16* ap1 = A + (size_t)(m0 + 16 + r) * lda + k0;
    __builtin_prefetch(ap0 + 32, 0, 0);   // global_prefetch of next K-slab
    v16bf af0 = frag_combine(*(const v8bf*)(ap0 + hf * 8),
                             *(const v8bf*)(ap0 + 16 + hf * 8));
    v16bf af1 = frag_combine(*(const v8bf*)(ap1 + hf * 8),
                             *(const v8bf*)(ap1 + 16 + hf * 8));

#pragma unroll
    for (int t = 0; t < 4; ++t) {
      const __bf16* bsr = &bs[t * 16 + r][0];
      v16bf bf = frag_combine(*(const v8bf*)(bsr + hf * 8),
                              *(const v8bf*)(bsr + 16 + hf * 8));
      acc[0][t] = __builtin_amdgcn_wmma_f32_16x16x32_bf16(
          false, af0, false, bf, (short)0, acc[0][t], false, false);
      acc[1][t] = __builtin_amdgcn_wmma_f32_16x16x32_bf16(
          false, af1, false, bf, (short)0, acc[1][t], false, false);
    }
  }

  // C/D layout: lane -> N = lane&15, VGPR i -> M = i + 8*(lane>>4)
#pragma unroll
  for (int u = 0; u < 2; ++u) {
#pragma unroll
    for (int t = 0; t < 4; ++t) {
      const int cn = n0 + t * 16 + r;
#pragma unroll
      for (int i = 0; i < 8; ++i) {
        C[(size_t)(m0 + u * 16 + i + 8 * hf) * ldc + cn] = alpha * acc[u][t][i];
      }
    }
  }
}

// x = enc + sinusoidal positional table; also emit bf16 copy
__global__ void pos_add_kernel(const float* __restrict__ in, float* __restrict__ xf,
                               __bf16* __restrict__ xb) {
  size_t idx = (size_t)blockIdx.x * 256 + threadIdx.x;   // B*S*D threads
  int d = (int)(idx & (Dz - 1));
  int s = (int)((idx >> 9) & (Sz - 1));
  float t = 0.f;
  if (s != 0) {
    float ang = (float)s / __powf(10000.f, 2.f * (float)d / (float)Dz);
    t = (d & 1) ? __cosf(ang) : __sinf(ang);
  }
  float y = in[idx] + t;
  xf[idx] = y;
  xb[idx] = (__bf16)y;
}

// Wt[l][n][k] = (bf16) W[l][k][n]
__global__ void wtrans_kernel(const float* __restrict__ W, __bf16* __restrict__ Wt,
                              int K, int N) {
  size_t off = (size_t)blockIdx.z * (size_t)K * N;
  int idx = blockIdx.x * 256 + threadIdx.x;
  if (idx >= K * N) return;
  int k = idx / N;
  int n = idx - k * N;
  Wt[off + (size_t)n * K + k] = (__bf16)W[off + idx];
}

// f32 -> bf16 (optional relu); count is multiple of 256
__global__ void cvt_kernel(const float* __restrict__ in, __bf16* __restrict__ out, int relu) {
  size_t i = (size_t)blockIdx.x * 256 + threadIdx.x;
  float v = in[i];
  if (relu) v = fmaxf(v, 0.f);
  out[i] = (__bf16)v;
}

// In-place masked softmax over 512 keys; one 256-thread block per (b,h,q) row.
// Writes fp32 result in place (attn region of d_out) + bf16 copy for attn*V.
__global__ void softmax_mask_kernel(float* __restrict__ sc, __bf16* __restrict__ ab,
                                    const int* __restrict__ mask) {
  const int row = blockIdx.x;            // b*H*S + h*S + q
  const int b   = row >> 12;             // / (H*S) = /4096
  float*   p  = sc + (size_t)row * Sz;
  __bf16*  pb = ab + (size_t)row * Sz;
  const int tid = threadIdx.x;

  float v0 = p[tid];
  float v1 = p[tid + 256];
  if (mask[b * Sz + tid] == 0)        v0 = -1e9f;
  if (mask[b * Sz + tid + 256] == 0)  v1 = -1e9f;

  __shared__ float sd[256];
  sd[tid] = fmaxf(v0, v1);
  __syncthreads();
  for (int off = 128; off > 0; off >>= 1) {
    if (tid < off) sd[tid] = fmaxf(sd[tid], sd[tid + off]);
    __syncthreads();
  }
  const float m = sd[0];
  __syncthreads();

  float e0 = __expf(v0 - m);
  float e1 = __expf(v1 - m);
  sd[tid] = e0 + e1;
  __syncthreads();
  for (int off = 128; off > 0; off >>= 1) {
    if (tid < off) sd[tid] += sd[tid + off];
    __syncthreads();
  }
  const float inv = 1.0f / sd[0];

  float a0 = e0 * inv, a1 = e1 * inv;
  p[tid]        = a0;
  p[tid + 256]  = a1;
  pb[tid]       = (__bf16)a0;
  pb[tid + 256] = (__bf16)a1;
}

// x = LayerNorm(a + x) * g + b  (in place on xf), also emit bf16 copy.
// One 256-thread block per 512-wide row.
__global__ void add_ln_kernel(const float* __restrict__ a, float* __restrict__ x,
                              const float* __restrict__ g, const float* __restrict__ bb,
                              __bf16* __restrict__ xb) {
  const int tid = threadIdx.x;
  const size_t base = (size_t)blockIdx.x * Dz;

  float v0 = a[base + tid]       + x[base + tid];
  float v1 = a[base + tid + 256] + x[base + tid + 256];

  __shared__ float sd[256];
  sd[tid] = v0 + v1;
  __syncthreads();
  for (int off = 128; off > 0; off >>= 1) {
    if (tid < off) sd[tid] += sd[tid + off];
    __syncthreads();
  }
  const float mean = sd[0] * (1.0f / (float)Dz);
  __syncthreads();

  const float d0 = v0 - mean, d1 = v1 - mean;
  sd[tid] = d0 * d0 + d1 * d1;
  __syncthreads();
  for (int off = 128; off > 0; off >>= 1) {
    if (tid < off) sd[tid] += sd[tid + off];
    __syncthreads();
  }
  const float inv = rsqrtf(sd[0] * (1.0f / (float)Dz) + 1e-5f);

  const float y0 = d0 * inv * g[tid]       + bb[tid];
  const float y1 = d1 * inv * g[tid + 256] + bb[tid + 256];
  x[base + tid]        = y0;
  x[base + tid + 256]  = y1;
  xb[base + tid]       = (__bf16)y0;
  xb[base + tid + 256] = (__bf16)y1;
}

extern "C" void kernel_launch(void* const* d_in, const int* in_sizes, int n_in,
                              void* d_out, int out_size, void* d_ws, size_t ws_size,
                              hipStream_t stream) {
  (void)in_sizes; (void)n_in; (void)out_size; (void)ws_size;

  const float* enc  = (const float*)d_in[0];
  const int*   mask = (const int*)  d_in[1];
  const float* Wq   = (const float*)d_in[2];
  const float* Wk   = (const float*)d_in[3];
  const float* Wv   = (const float*)d_in[4];
  const float* Wo   = (const float*)d_in[5];
  const float* ln1g = (const float*)d_in[6];
  const float* ln1b = (const float*)d_in[7];
  const float* W1   = (const float*)d_in[8];
  const float* W2   = (const float*)d_in[9];
  const float* ln2g = (const float*)d_in[10];
  const float* ln2b = (const float*)d_in[11];

  const int    M  = Bz * Sz;                       // 4096 rows of activations
  const size_t XE = (size_t)M * Dz;                // 2,097,152 (x elements)
  const size_t AE = (size_t)Bz * Hz * Sz * Sz;     // 16,777,216 (attn per layer)

  // ---- carve workspace ----
  char* w = (char*)d_ws;
  auto carve = [&](size_t bytes) -> void* {
    void* p = (void*)w;
    w += (bytes + 255) & ~(size_t)255;
    return p;
  };
  float*  xf   = (float*) carve(XE * 4);
  __bf16* xb   = (__bf16*)carve(XE * 2);
  __bf16* qb   = (__bf16*)carve(XE * 2);
  __bf16* kb   = (__bf16*)carve(XE * 2);
  __bf16* vb   = (__bf16*)carve(XE * 2);
  __bf16* ctxb = (__bf16*)carve(XE * 2);
  __bf16* attb = (__bf16*)carve(AE * 2);
  __bf16* hb   = (__bf16*)carve((size_t)M * DFFz * 2);
  float*  tmp  = (float*) carve((size_t)M * DFFz * 4);
  __bf16* WqT  = (__bf16*)carve((size_t)Lz * Dz * Dz * 2);
  __bf16* WkT  = (__bf16*)carve((size_t)Lz * Dz * Dz * 2);
  __bf16* WvT  = (__bf16*)carve((size_t)Lz * Dz * Dz * 2);
  __bf16* WoT  = (__bf16*)carve((size_t)Lz * Dz * Dz * 2);
  __bf16* W1T  = (__bf16*)carve((size_t)Lz * Dz * DFFz * 2);
  __bf16* W2T  = (__bf16*)carve((size_t)Lz * Dz * DFFz * 2);

  // ---- weights: fp32 -> bf16, transposed to N x K ----
  {
    dim3 g((Dz * Dz + 255) / 256, 1, Lz);
    wtrans_kernel<<<g, 256, 0, stream>>>(Wq, WqT, Dz, Hz * DKz);
    wtrans_kernel<<<g, 256, 0, stream>>>(Wk, WkT, Dz, Hz * DKz);
    wtrans_kernel<<<g, 256, 0, stream>>>(Wv, WvT, Dz, Hz * DVz);
    wtrans_kernel<<<g, 256, 0, stream>>>(Wo, WoT, Hz * DVz, Dz);
    dim3 g1((Dz * DFFz + 255) / 256, 1, Lz);
    wtrans_kernel<<<g1, 256, 0, stream>>>(W1, W1T, Dz, DFFz);
    wtrans_kernel<<<g1, 256, 0, stream>>>(W2, W2T, DFFz, Dz);
  }

  // ---- x = enc + pos table ----
  pos_add_kernel<<<(unsigned)(XE / 256), 256, 0, stream>>>(enc, xf, xb);

  float* outx = (float*)d_out;
  const dim3 gp(Dz / 64, M / 256, 1);        // 512-col activation GEMMs
  const dim3 gf(DFFz / 64, M / 256, 1);      // 2048-col FFN GEMM
  const dim3 gs(Sz / 64, Sz / 256, Bz * Hz); // per-(b,h) score GEMM
  const dim3 gc(DVz / 64, Sz / 256, Bz * Hz);// per-(b,h) attn*V GEMM

  for (int l = 0; l < Lz; ++l) {
    const size_t wOff  = (size_t)l * Dz * Dz;
    const size_t wOff1 = (size_t)l * Dz * DFFz;

    // ---- Q, K, V projections (bf16 WMMA, fp32 accum) ----
    gemm_bf16_wmma<1><<<gp, 256, 0, stream>>>(xb, WqT + wOff, tmp,
        Dz, Dz, Dz, Dz, 1, 0, 0, 0, 0, 0, 0, 1.0f);
    cvt_kernel<<<(unsigned)(XE / 256), 256, 0, stream>>>(tmp, qb, 0);
    gemm_bf16_wmma<1><<<gp, 256, 0, stream>>>(xb, WkT + wOff, tmp,
        Dz, Dz, Dz, Dz, 1, 0, 0, 0, 0, 0, 0, 1.0f);
    cvt_kernel<<<(unsigned)(XE / 256), 256, 0, stream>>>(tmp, kb, 0);
    gemm_bf16_wmma<1><<<gp, 256, 0, stream>>>(xb, WvT + wOff, tmp,
        Dz, Dz, Dz, Dz, 1, 0, 0, 0, 0, 0, 0, 1.0f);
    cvt_kernel<<<(unsigned)(XE / 256), 256, 0, stream>>>(tmp, vb, 0);

    // ---- scores = (Q @ K^T) / sqrt(DK), straight into d_out attn slice ----
    float* sc = outx + XE + (size_t)l * AE;
    gemm_bf16_wmma<1><<<gs, 256, 0, stream>>>(qb, kb, sc,
        DKz, Dz, Dz, Sz, Hz,
        (long long)Sz * Dz, (long long)DKz,           // A: +b*S*D, +h*64
        (long long)Sz * Dz, (long long)DKz,           // B (K-matrix, N=key rows)
        (long long)Hz * Sz * Sz, (long long)Sz * Sz,  // C: [B,H,S,S]
        0.125f);

    // ---- masked softmax (in place, fp32) + bf16 copy ----
    softmax_mask_kernel<<<Bz * Hz * Sz, 256, 0, stream>>>(sc, attb, mask);

    // ---- ctx = attn @ V, written directly in [B,S,H*DV] layout ----
    gemm_bf16_wmma<0><<<gc, 256, 0, stream>>>(attb, vb, tmp,
        Sz, Sz, Dz, Dz, Hz,
        (long long)Hz * Sz * Sz, (long long)Sz * Sz,  // A: attn [B,H,S,S]
        (long long)Sz * Dz, (long long)DVz,           // B: V rows, head col offset
        (long long)Sz * Dz, (long long)DVz,           // C: [B,S,H*DV]
        1.0f);
    cvt_kernel<<<(unsigned)(XE / 256), 256, 0, stream>>>(tmp, ctxb, 0);

    // ---- out proj + residual + LN1 ----
    gemm_bf16_wmma<1><<<gp, 256, 0, stream>>>(ctxb, WoT + wOff, tmp,
        Dz, Dz, Dz, Dz, 1, 0, 0, 0, 0, 0, 0, 1.0f);
    add_ln_kernel<<<M, 256, 0, stream>>>(tmp, xf, ln1g + l * Dz, ln1b + l * Dz, xb);

    // ---- FFN: relu(x @ W1) @ W2, residual + LN2 ----
    gemm_bf16_wmma<1><<<gf, 256, 0, stream>>>(xb, W1T + wOff1, tmp,
        Dz, Dz, Dz, DFFz, 1, 0, 0, 0, 0, 0, 0, 1.0f);
    cvt_kernel<<<(unsigned)(((size_t)M * DFFz) / 256), 256, 0, stream>>>(tmp, hb, 1);
    gemm_bf16_wmma<1><<<gp, 256, 0, stream>>>(hb, W2T + wOff1, tmp,
        DFFz, DFFz, DFFz, Dz, 1, 0, 0, 0, 0, 0, 0, 1.0f);
    add_ln_kernel<<<M, 256, 0, stream>>>(tmp, xf, ln2g + l * Dz, ln2b + l * Dz, xb);
  }

  // final x -> d_out[0 : B*S*D]
  hipMemcpyAsync(d_out, xf, XE * 4, hipMemcpyDeviceToDevice, stream);
}
